// RAM_21809843929536
// MI455X (gfx1250) — matrix operand order repaired and checked by reading, not compiled, not run on hardware
//
#include <hip/hip_runtime.h>

#define B_   4
#define C_   256
#define HID  32
#define N_   4096
#define SCALE 0.17677669529663687f   // 1/sqrt(32)

typedef __attribute__((ext_vector_type(16))) __bf16 v16bf;
typedef __attribute__((ext_vector_type(8)))  __bf16 v8bf;
typedef __attribute__((ext_vector_type(8)))  float  v8f;

__device__ __forceinline__ unsigned short f2bf(float f) {
    unsigned u = __float_as_uint(f);
    u += 0x7FFFu + ((u >> 16) & 1u);     // round-to-nearest-even
    return (unsigned short)(u >> 16);
}
__device__ __forceinline__ v16bf cat8(v8bf lo, v8bf hi) {
    return __builtin_shufflevector(lo, hi, 0,1,2,3,4,5,6,7,8,9,10,11,12,13,14,15);
}
__device__ __forceinline__ v8f wmma_bf16(v16bf a, v16bf b, v8f c) {
    return __builtin_amdgcn_wmma_f32_16x16x32_bf16(false, a, false, b, (short)0, c, false, false);
}

#define LSTRIDE 40   // halves; keeps 16B alignment, bank-friendly

// Stage a 32c x 64n f32 tile of x into LDS, transposed+converted: Lt[n][c] bf16.
// Each lane: one float4 (coalesced 16B over n), 4 bf16 scatter stores. 2 passes.
__device__ __forceinline__ void stage_xt(const float* __restrict__ x, size_t rowbase,
                                         int c0, int n0, unsigned short* Lt, int t) {
    #pragma unroll
    for (int pass = 0; pass < 2; pass++) {
        int ci = (t >> 4) + pass * 16;      // 0..31
        int nj = (t & 15) * 4;              // 0..60
        const float4 v = *reinterpret_cast<const float4*>(
            &x[(rowbase + (size_t)(c0 + ci)) * N_ + n0 + nj]);
        Lt[(nj + 0) * LSTRIDE + ci] = f2bf(v.x);
        Lt[(nj + 1) * LSTRIDE + ci] = f2bf(v.y);
        Lt[(nj + 2) * LSTRIDE + ci] = f2bf(v.z);
        Lt[(nj + 3) * LSTRIDE + ci] = f2bf(v.w);
    }
}

// ---------------- f32 -> bf16 weight conversion ----------------
__global__ void cvt_bf16(const float* __restrict__ src, unsigned short* __restrict__ dst, int n) {
    int i = blockIdx.x * 256 + threadIdx.x;
    if (i < n) dst[i] = f2bf(src[i]);
}

// ------- Projection (key/query): out[b][n][32] = x^T @ w^T + bias, all-WMMA -------
__global__ void __launch_bounds__(256) proj_kq(
        const float* __restrict__ x, const unsigned short* __restrict__ wb,
        const float* __restrict__ bias, unsigned short* __restrict__ out) {
    __shared__ unsigned short Lt[64 * LSTRIDE];   // [n_local][c_local] bf16
    int b  = blockIdx.x >> 6;
    int n0 = (blockIdx.x & 63) << 6;
    int t  = threadIdx.x;
    int w  = t >> 5, lane = t & 31;
    int col = lane & 15, hi = lane >> 4;
    int nt = w & 3;         // n sub-tile 0..3
    int ht = w >> 2;        // h sub-tile 0..1
    v8f acc = {};
    for (int c0 = 0; c0 < C_; c0 += 32) {
        __syncthreads();
        stage_xt(x, (size_t)b * C_, c0, n0, Lt, t);
        __syncthreads();
        const v8bf* ar = (const v8bf*)&Lt[(nt * 16 + col) * LSTRIDE];
        v16bf afrag = cat8(ar[2 * hi], ar[2 * hi + 1]);
        const v8bf* br = (const v8bf*)(wb + (size_t)(ht * 16 + col) * C_ + c0 + 16 * hi);
        v16bf bfrag = cat8(br[0], br[1]);
        acc = wmma_bf16(afrag, bfrag, acc);
    }
    int h = ht * 16 + col;
    float bv = bias[h];
    #pragma unroll
    for (int i = 0; i < 8; i++) {
        int n = n0 + nt * 16 + i + 8 * hi;
        out[((size_t)b * N_ + n) * HID + h] = f2bf(acc[i] + bv);
    }
}

// ------- Projection (val): out[b][o][n] = w2 @ x + b2, all-WMMA -------
__global__ void __launch_bounds__(256) proj_val(
        const float* __restrict__ x, const unsigned short* __restrict__ w2b,
        const float* __restrict__ bias, unsigned short* __restrict__ out) {
    __shared__ unsigned short Lt[64 * LSTRIDE];   // [n_local][c_local] bf16
    int b  = blockIdx.x >> 6;
    int n0 = (blockIdx.x & 63) << 6;
    int t  = threadIdx.x;
    int w  = t >> 5, lane = t & 31;
    int col = lane & 15, hi = lane >> 4;
    int c_base = w << 5;    // 32 output channels per wave
    v8f acc[2][4];
    #pragma unroll
    for (int r = 0; r < 2; r++)
        #pragma unroll
        for (int j = 0; j < 4; j++) acc[r][j] = {};
    for (int c0 = 0; c0 < C_; c0 += 32) {
        __syncthreads();
        stage_xt(x, (size_t)b * C_, c0, n0, Lt, t);
        __syncthreads();
        const v8bf* ar0 = (const v8bf*)(w2b + (size_t)(c_base + col) * C_ + c0);
        const v8bf* ar1 = (const v8bf*)(w2b + (size_t)(c_base + 16 + col) * C_ + c0);
        v16bf aw0 = cat8(ar0[hi], ar0[hi + 2]);
        v16bf aw1 = cat8(ar1[hi], ar1[hi + 2]);
        #pragma unroll
        for (int j = 0; j < 4; j++) {
            const v8bf* pr = (const v8bf*)&Lt[(j * 16 + col) * LSTRIDE + 8 * hi];
            v16bf bfrag = cat8(pr[0], pr[2]);
            acc[0][j] = wmma_bf16(aw0, bfrag, acc[0][j]);
            acc[1][j] = wmma_bf16(aw1, bfrag, acc[1][j]);
        }
    }
    #pragma unroll
    for (int r = 0; r < 2; r++)
        #pragma unroll
        for (int j = 0; j < 4; j++) {
            #pragma unroll
            for (int i = 0; i < 8; i++) {
                int o = c_base + r * 16 + i + 8 * hi;
                int n = n0 + j * 16 + col;
                out[((size_t)b * C_ + o) * N_ + n] = f2bf(acc[r][j][i] + bias[o]);
            }
        }
}

// ---------------- Pass 1: per-row softmax stats (online max/sum over all m) ----------------
__global__ void row_stats(const unsigned short* __restrict__ key,
                          const unsigned short* __restrict__ qt,
                          float* __restrict__ rowM, float* __restrict__ rowZ) {
    int wave = (int)((blockIdx.x * blockDim.x + threadIdx.x) >> 5);  // one 16-row tile per wave
    int lane = threadIdx.x & 31;
    int b  = wave >> 8;
    int n0 = (wave & 255) << 4;
    int col = lane & 15, hi = lane >> 4;

    const v8bf* krow = (const v8bf*)(key + ((size_t)b * N_ + n0 + col) * HID);
    v16bf afrag = cat8(krow[hi], krow[hi + 2]);

    float m[8], z[8];
    #pragma unroll
    for (int i = 0; i < 8; i++) { m[i] = -INFINITY; z[i] = 0.f; }

    const unsigned short* qb = qt + (size_t)b * N_ * HID;
    for (int mt = 0; mt < N_ / 16; mt++) {
        const v8bf* qrow = (const v8bf*)(qb + (size_t)(mt * 16 + col) * HID);
        v16bf bfrag = cat8(qrow[hi], qrow[hi + 2]);
        v8f s = {};
        s = wmma_bf16(afrag, bfrag, s);
        #pragma unroll
        for (int i = 0; i < 8; i++) {
            float sv = s[i] * SCALE;
            float nm = fmaxf(m[i], sv);
            z[i] = z[i] * __expf(m[i] - nm) + __expf(sv - nm);
            m[i] = nm;
        }
    }
    #pragma unroll
    for (int off = 8; off > 0; off >>= 1) {
        #pragma unroll
        for (int i = 0; i < 8; i++) {
            float om = __shfl_xor(m[i], off, 16);
            float oz = __shfl_xor(z[i], off, 16);
            float nm = fmaxf(m[i], om);
            z[i] = z[i] * __expf(m[i] - nm) + oz * __expf(om - nm);
            m[i] = nm;
        }
    }
    if (col == 0) {
        #pragma unroll
        for (int i = 0; i < 8; i++) {
            int r = n0 + i + 8 * hi;
            rowM[(size_t)b * N_ + r] = m[i];
            rowZ[(size_t)b * N_ + r] = z[i];
        }
    }
}

// ---------------- Pass 2: out[:,m_tile] = val @ softmax-normalized P ----------------
#define PSTRIDE 40
__global__ void __launch_bounds__(256) attn_pv(
        const unsigned short* __restrict__ key, const unsigned short* __restrict__ qt,
        const unsigned short* __restrict__ val,
        const float* __restrict__ rowM, const float* __restrict__ rowZ,
        float* __restrict__ out) {
    __shared__ unsigned short P_t[64 * PSTRIDE];  // [m_local][n_local] bf16
    __shared__ float m_s[32], zinv_s[32];

    int b  = blockIdx.x >> 6;
    int m0 = (blockIdx.x & 63) << 6;
    int t  = threadIdx.x;
    int w  = t >> 5, lane = t & 31;
    int col = lane & 15, hi = lane >> 4;
    int mj    = w & 3;
    int n_sub = ((w >> 2) << 4);
    int c_base = w << 5;

    const unsigned short* qb   = qt  + (size_t)b * N_ * HID;
    const unsigned short* keyb = key + (size_t)b * N_ * HID;
    const unsigned short* valb = val + (size_t)b * C_ * (size_t)N_;
    const float* rmb = rowM + (size_t)b * N_;
    const float* rzb = rowZ + (size_t)b * N_;

    const v8bf* qrow = (const v8bf*)(qb + (size_t)(m0 + mj * 16 + col) * HID);
    v16bf qfrag = cat8(qrow[hi], qrow[hi + 2]);

    v8f acc[2][4];
    #pragma unroll
    for (int r = 0; r < 2; r++)
        #pragma unroll
        for (int j = 0; j < 4; j++) acc[r][j] = {};

    for (int nc = 0; nc < N_; nc += 32) {
        if (t < 32) {
            m_s[t]    = rmb[nc + t];
            zinv_s[t] = 1.0f / rzb[nc + t];
        }
        const v8bf* krow = (const v8bf*)(keyb + (size_t)(nc + n_sub + col) * HID);
        v16bf afrag = cat8(krow[hi], krow[hi + 2]);
        v8f s = {};
        s = wmma_bf16(afrag, qfrag, s);
        __syncthreads();
        #pragma unroll
        for (int i = 0; i < 8; i += 2) {
            int r = n_sub + i + 8 * hi;
            float p0 = __expf(s[i]     * SCALE - m_s[r])     * zinv_s[r];
            float p1 = __expf(s[i + 1] * SCALE - m_s[r + 1]) * zinv_s[r + 1];
            unsigned pk = (unsigned)f2bf(p0) | ((unsigned)f2bf(p1) << 16);
            *(unsigned*)&P_t[(mj * 16 + col) * PSTRIDE + r] = pk;
        }
        __syncthreads();
        const v8bf* vr0 = (const v8bf*)(valb + (size_t)(c_base + col) * N_ + nc);
        const v8bf* vr1 = (const v8bf*)(valb + (size_t)(c_base + 16 + col) * N_ + nc);
        v16bf av0 = cat8(vr0[hi], vr0[hi + 2]);
        v16bf av1 = cat8(vr1[hi], vr1[hi + 2]);
        #pragma unroll
        for (int j = 0; j < 4; j++) {
            const v8bf* pr = (const v8bf*)&P_t[(j * 16 + col) * PSTRIDE + hi * 8];
            v16bf bfrag = cat8(pr[0], pr[2]);
            acc[0][j] = wmma_bf16(av0, bfrag, acc[0][j]);
            acc[1][j] = wmma_bf16(av1, bfrag, acc[1][j]);
        }
    }
    #pragma unroll
    for (int r = 0; r < 2; r++)
        #pragma unroll
        for (int j = 0; j < 4; j++)
            #pragma unroll
            for (int i = 0; i < 8; i++) {
                int c = c_base + r * 16 + i + 8 * hi;
                int mcol = m0 + j * 16 + col;
                out[((size_t)b * C_ + c) * N_ + mcol] = acc[r][j][i];
            }
}

extern "C" void kernel_launch(void* const* d_in, const int* in_sizes, int n_in,
                              void* d_out, int out_size, void* d_ws, size_t ws_size,
                              hipStream_t stream) {
    (void)in_sizes; (void)n_in; (void)out_size; (void)ws_size;
    const float* left = (const float*)d_in[0];
    const float* up   = (const float*)d_in[1];
    const float* w0   = (const float*)d_in[2];
    const float* b0   = (const float*)d_in[3];
    const float* w1   = (const float*)d_in[4];
    const float* b1   = (const float*)d_in[5];
    const float* w2   = (const float*)d_in[6];
    const float* b2   = (const float*)d_in[7];
    float* out = (float*)d_out;

    char* ws = (char*)d_ws;
    unsigned short* key  = (unsigned short*)ws;                        // 1 MB  [B][N][32] bf16
    unsigned short* qt   = (unsigned short*)(ws + (1u << 20));         // 1 MB  [B][M][32] bf16
    unsigned short* val  = (unsigned short*)(ws + (2u << 20));         // 8 MB  [B][C][N]  bf16
    float* rowM          = (float*)(ws + (10u << 20));                 // 64 KB
    float* rowZ          = (float*)(ws + (10u << 20) + (1u << 16));    // 64 KB
    unsigned short* w0b  = (unsigned short*)(ws + (10u << 20) + (2u << 16));            // 16 KB
    unsigned short* w1b  = (unsigned short*)(ws + (10u << 20) + (2u << 16) + 16384);    // 16 KB
    unsigned short* w2b  = (unsigned short*)(ws + (10u << 20) + (2u << 16) + 32768);    // 128 KB

    cvt_bf16<<<(HID * C_ + 255) / 256, 256, 0, stream>>>(w0, w0b, HID * C_);
    cvt_bf16<<<(HID * C_ + 255) / 256, 256, 0, stream>>>(w1, w1b, HID * C_);
    cvt_bf16<<<(C_ * C_ + 255) / 256, 256, 0, stream>>>(w2, w2b, C_ * C_);

    proj_kq <<<B_ * (N_ / 64), 256, 0, stream>>>(left, w0b, b0, key);
    proj_kq <<<B_ * (N_ / 64), 256, 0, stream>>>(up,   w1b, b1, qt);
    proj_val<<<B_ * (N_ / 64), 256, 0, stream>>>(left, w2b, b2, val);
    row_stats<<<(B_ * (N_ / 16)) / 8, 256, 0, stream>>>(key, qt, rowM, rowZ);
    attn_pv  <<<B_ * (N_ / 64), 256, 0, stream>>>(key, qt, val, rowM, rowZ, out);
}
